// CTSimGLM_26448408609572
// MI455X (gfx1250) — matrix operand order repaired and checked by reading, not compiled, not run on hardware
//
#include <hip/hip_runtime.h>

typedef _Float16 v16h __attribute__((ext_vector_type(16)));
typedef _Float16 v8h  __attribute__((ext_vector_type(8)));
typedef float    v8f  __attribute__((ext_vector_type(8)));

#if __has_builtin(__builtin_amdgcn_sched_barrier)
#define SCHED_FENCE() __builtin_amdgcn_sched_barrier(0)
#else
#define SCHED_FENCE()
#endif

#define B_    4
#define P_    4096
#define T_    2000
#define K_    250
#define C_    24
#define R_    32
#define TSIM  1750      // L-1 = (T-K+1)-1
#define GPAD  1792      // padded gensig stride per batch
#define NBLK  110       // ceil(TSIM/16)

// ---------------------------------------------------------------------------
// Kernel 1: spatial projection  spat[b,t] = sum_p sf[p] * stim[b,p,t]
// Streams the full 131 MB stimulus once with b128 loads (4 t per thread).
// ---------------------------------------------------------------------------
__global__ void __launch_bounds__(256)
glm_spat(const float* __restrict__ stim, const float* __restrict__ sf,
         float* __restrict__ spat) {
  __shared__ float s_sf[P_];
  int tid = threadIdx.x;
  for (int i = tid; i < P_; i += 256) s_sf[i] = sf[i];
  __syncthreads();
  int b = blockIdx.x >> 1;                      // 2 tiles per batch
  int q = (blockIdx.x & 1) * 256 + tid;         // t-quad index, 500 quads
  if (q >= T_ / 4) return;
  const float4* sp = (const float4*)(stim + (size_t)b * P_ * T_) + q;
  float4 acc = make_float4(0.f, 0.f, 0.f, 0.f);
#pragma unroll 4
  for (int p = 0; p < P_; ++p) {
    float4 v = sp[(size_t)p * (T_ / 4)];
    float s = s_sf[p];
    acc.x += s * v.x; acc.y += s * v.y; acc.z += s * v.z; acc.w += s * v.w;
  }
  ((float4*)(spat + b * T_))[q] = acc;
}

// ---------------------------------------------------------------------------
// Kernel 2: gensig[b,s] = bias + conv(spat, tc)[s] + conv(cs, cf)[s]
// ---------------------------------------------------------------------------
__global__ void __launch_bounds__(256)
glm_gensig(const float* __restrict__ spat, const float* __restrict__ tc,
           const float* __restrict__ cf, const float* __restrict__ cs,
           const float* __restrict__ bias, float* __restrict__ gsig) {
  __shared__ float s_tc[K_];
  __shared__ float s_cf[C_ * K_];
  int tid = threadIdx.x;
  for (int i = tid; i < K_; i += 256)      s_tc[i] = tc[i];
  for (int i = tid; i < C_ * K_; i += 256) s_cf[i] = cf[i];
  __syncthreads();
  int b = blockIdx.x / 7;
  int s = (blockIdx.x % 7) * 256 + tid;
  if (s >= TSIM) return;
  float acc = bias[0];
  const float* spb = spat + b * T_ + s;
#pragma unroll 2
  for (int k = 0; k < K_; ++k) acc += s_tc[k] * spb[k];
  for (int c = 0; c < C_; ++c) {
    const float* csb = cs + ((size_t)b * C_ + c) * T_ + s;
    const float* w = s_cf + c * K_;
#pragma unroll 2
    for (int k = 0; k < K_; ++k) acc += w[k] * csb[k];
  }
  gsig[b * GPAD + s] = acc;
}

// ---------------------------------------------------------------------------
// Kernel 3: blocked sigmoid-rate recurrence.
// 8 single-wave blocks, 16 streams each. Per 16-step time block:
//   OldFB^T (16 streams x 16 steps) = H^T(16x256) * F'(256x16)  via
//   8x v_wmma_f32_16x16x32_f16, software-pipelined with sched_barriers so
//   fragment kk+1's ds_load_b128 pair overlaps WMMA kk (partial dscnt waits),
//   then a sequential in-block correction with register-resident taps
//   and a fast v_rcp-based sigmoid.
// Mirrored f16 history ring (512 slots/stream): contiguous 16B-aligned runs.
// ---------------------------------------------------------------------------
__global__ void __launch_bounds__(32)
glm_sim(const float* __restrict__ init, const float* __restrict__ ff,
        const float* __restrict__ gsig, float* __restrict__ out) {
  __shared__ __attribute__((aligned(16))) _Float16 s_hist[16 * 512]; // [stream][mirrored ring]
  __shared__ float s_fbf[256];          // feedback filter, zero padded
  __shared__ float s_oldfb[16 * 16];    // [i][local stream]

  const int lane  = threadIdx.x;        // 0..31 (wave32)
  const int blk   = blockIdx.x;         // 0..7 -> streams blk*16..blk*16+15
  const int bcell = (blk * 16) >> 5;    // batch index (16-aligned range never crosses R=32)

  for (int k = lane; k < 256; k += 32) s_fbf[k] = (k < K_) ? ff[k] : 0.f;

  // init ring with win0 (out times 0..249 -> slots 0..249; 250..255 = 0),
  // mirrored at +256; also write the initial 250 outputs per stream.
  for (int jl = 0; jl < 16; ++jl) {
    int j = blk * 16 + jl;
    const float* ib = init + bcell * K_;
    for (int s = lane; s < 256; s += 32) {
      float v = (s < K_) ? ib[s] : 0.f;
      _Float16 h = (_Float16)v;
      s_hist[jl * 512 + s]       = h;
      s_hist[jl * 512 + s + 256] = h;
      if (s < K_) out[(size_t)j * T_ + s] = v;
    }
  }
  __syncthreads();

  // Constant B fragments: F'[u,n] = (u >= n && u <= 249) ? f[u-n] : 0
  // (columns 250..255 zeroed: they alias not-yet-computed in-block spikes)
  // ISA 16-bit B 32x16 layout: n = lane&15; lanes<16 -> K {0..7,16..23}, lanes>=16 -> K {8..15,24..31}.
  const int nB  = lane & 15;
  const int kbB = (lane < 16) ? 0 : 8;
  v16h bfr[8];
#pragma unroll
  for (int kk = 0; kk < 8; ++kk) {
    v16h bf;
#pragma unroll
    for (int h = 0; h < 16; ++h) {
      int u = 32 * kk + kbB + h + (h >= 8 ? 8 : 0);
      bf[h] = (u >= nB && u <= 249) ? (_Float16)s_fbf[u - nB] : (_Float16)0.f;
    }
    bfr[kk] = bf;
  }

  // Last 15 feedback taps f[235..249] in registers for the in-block correction:
  // coefficient f[250-i+m] == ftap[15-i+m].
  float ftap[15];
#pragma unroll
  for (int d = 0; d < 15; ++d) ftap[d] = s_fbf[235 + d];

  // ISA 16-bit A 16x32 layout: m = lane&15; lanes<16 -> K {0..7,16..23}, lanes>=16 -> K {8..15,24..31}
  const int mA  = lane & 15;
  const int kbA = (lane < 16) ? 0 : 8;
  const float* gb = gsig + bcell * GPAD;

  for (int nb = 0; nb < NBLK; ++nb) {
    const int s0 = nb * 16;
    int rem = TSIM - s0; if (rem > 16) rem = 16;
    const int ringbase = s0 & 255;            // multiple of 16 -> 16B-aligned runs

    // ---- GEMM phase: all 32 lanes, EXEC all ones ----
    // Software pipeline: loads for kk+1 are fenced ABOVE wmma kk, so the two
    // outstanding ds_load_b128 overlap the matrix op (wait dscnt <= 2).
    const _Float16* row = s_hist + mA * 512 + ringbase + kbA;
    v8h lo[2], hi[2];
    lo[0] = *(const v8h*)(row);          // ds_load_b128
    hi[0] = *(const v8h*)(row + 16);     // ds_load_b128
    v8f cacc = {};
#pragma unroll
    for (int kk = 0; kk < 8; ++kk) {
      const int cur = kk & 1, nxt = cur ^ 1;
      if (kk < 7) {
        lo[nxt] = *(const v8h*)(row + 32 * (kk + 1));
        hi[nxt] = *(const v8h*)(row + 32 * (kk + 1) + 16);
      }
      SCHED_FENCE();   // loads above may not sink below the wmma
      v16h a = __builtin_shufflevector(lo[cur], hi[cur], 0, 1, 2, 3, 4, 5, 6, 7,
                                                         8, 9, 10, 11, 12, 13, 14, 15);
      cacc = __builtin_amdgcn_wmma_f32_16x16x32_f16(false, a, false, bfr[kk],
                                                    (short)0, cacc, false, false);
      SCHED_FENCE();   // wmma may not hoist above its fence
    }

    // C/D layout: n = lane&15 (= step i), m = vgpr + (lane<16 ? 0 : 8) (= local stream)
    {
      const int n    = lane & 15;
      const int moff = (lane < 16) ? 0 : 8;
#pragma unroll
      for (int v = 0; v < 8; ++v)
        s_oldfb[n * 16 + (v + moff)] = cacc[v];
    }
    __syncthreads();   // single-wave workgroup: cheap

    // ---- sequential phase: lanes 0..15 own one stream each ----
    if (lane < 16) {
      const int j = blk * 16 + lane;
      float sblk[16];
#pragma unroll
      for (int i = 0; i < 16; ++i) sblk[i] = 0.f;
#pragma unroll
      for (int i = 0; i < 16; ++i) {
        float fb = s_oldfb[i * 16 + lane];
#pragma unroll
        for (int m = 0; m < i; ++m) fb += ftap[15 - i + m] * sblk[m];
        float x = gb[s0 + i] + fb;               // padded gsig: safe for i >= rem
        // fast sigmoid: v_exp_f32 + v_add + v_rcp_f32 (no IEEE divide sequence)
        float spike = __builtin_amdgcn_rcpf(1.f + __expf(-x));
        if (i < rem) {
          sblk[i] = spike;
          const int tnew = K_ + s0 + i;          // absolute output time
          out[(size_t)j * T_ + tnew] = spike;
          const int slot = tnew & 255;
          _Float16 hs = (_Float16)spike;
          s_hist[lane * 512 + slot]       = hs;  // mirrored write
          s_hist[lane * 512 + slot + 256] = hs;
        }
      }
    }
    __syncthreads();
  }
}

// ---------------------------------------------------------------------------
extern "C" void kernel_launch(void* const* d_in, const int* in_sizes, int n_in,
                              void* d_out, int out_size, void* d_ws, size_t ws_size,
                              hipStream_t stream) {
  const float* stim = (const float*)d_in[0];   // (B,P,T)
  const float* init = (const float*)d_in[1];   // (B,K)
  const float* cs   = (const float*)d_in[2];   // (B,C,T)
  const float* sf   = (const float*)d_in[3];   // (P,)
  const float* tc   = (const float*)d_in[4];   // (K,)
  const float* ff   = (const float*)d_in[5];   // (K,)
  const float* cf   = (const float*)d_in[6];   // (C,K)
  const float* bias = (const float*)d_in[7];   // (1,)
  // d_in[8] = n_repeats (compile-time R_=32 per reference setup)

  float* out  = (float*)d_out;                 // (B,R,T)
  float* spat = (float*)d_ws;                  // B*T floats
  float* gsig = spat + B_ * T_;                // B*GPAD floats

  glm_spat  <<<B_ * 2, 256, 0, stream>>>(stim, sf, spat);
  glm_gensig<<<B_ * 7, 256, 0, stream>>>(spat, tc, cf, cs, bias, gsig);
  glm_sim   <<<8,      32,  0, stream>>>(init, ff, gsig, out);
}
